// Encoder_5729486373367
// MI455X (gfx1250) — matrix operand order repaired and checked by reading, not compiled, not run on hardware
//
#include <hip/hip_runtime.h>
#include <cstdint>

// ---------------- problem constants ----------------
#define L_   6
#define D_   512
#define H_   8
#define DFF_ 2048
#define B_   4
#define N_   4096
#define HD_  64
#define M_   (B_ * N_)   // 16384 rows

typedef _Float16 h16;
typedef __attribute__((ext_vector_type(16))) _Float16 v16h;
typedef __attribute__((ext_vector_type(8)))  _Float16 v8h;
typedef __attribute__((ext_vector_type(4)))  _Float16 v4h;
typedef __attribute__((ext_vector_type(8)))  float    v8f;
typedef int v4i __attribute__((__vector_size__(16)));
typedef unsigned int v4u __attribute__((__vector_size__(16)));
typedef int v8i __attribute__((__vector_size__(32)));

#define AS1_ __attribute__((address_space(1)))
#define AS3_ __attribute__((address_space(3)))

__device__ __forceinline__ v16h cat16(v8h lo, v8h hi) {
  return __builtin_shufflevector(lo, hi, 0,1,2,3,4,5,6,7,8,9,10,11,12,13,14,15);
}
__device__ __forceinline__ v8f wmma16(v16h a, v16h b, v8f c) {
  // D(f32 16x16) = A(f16 16x32) * B(f16 32x16) + C  -- wave32 CDNA5 WMMA
  return __builtin_amdgcn_wmma_f32_16x16x32_f16(false, a, false, b, (short)0, c, false, false);
}
__device__ __forceinline__ float elu1(float x) { return x > 0.0f ? x : expm1f(x); }

// ---- CDNA5 async global->LDS copy (GLOBAL_LOAD_ASYNC_TO_LDS_B128, ASYNCcnt) ----
#if __has_builtin(__builtin_amdgcn_global_load_async_to_lds_b128)
#define HAVE_ASYNC_LDS 1
__device__ __forceinline__ void async_cp16B(const void* gsrc, const void* ldst) {
  // flat global address == AS1 address; LDS generic address truncates to AS3 offset (ISA 10.2)
  __builtin_amdgcn_global_load_async_to_lds_b128(
      (AS1_ v4i*)(uintptr_t)gsrc,
      (AS3_ v4i*)(unsigned int)(uintptr_t)ldst,
      0, 0);
}
#else
#define HAVE_ASYNC_LDS 0
#endif

__device__ __forceinline__ void wait_async0() {
#if __has_builtin(__builtin_amdgcn_s_wait_asynccnt)
  __builtin_amdgcn_s_wait_asynccnt(0);
#else
  asm volatile("s_wait_asynccnt 0" ::: "memory");
#endif
}

// ---- CDNA5 Tensor Data Mover: one instruction stages a whole 2D tile to LDS ----
#if __has_builtin(__builtin_amdgcn_tensor_load_to_lds)
#define HAVE_TDM 1
// D# packing per ISA 8.3-8.5. 2D tile of 16-bit elements, LDS padding to avoid
// bank conflicts (pad_interval: DWORDs-per-row, pad_amount: pad DWORDs-1).
// Toolchain probe: 6-arg form (g0, g1, g2, g3, extra int32x8, cpol).
__device__ __forceinline__ void tdm_load_2d(const void* gaddr, const void* laddr,
                                            int tensor_d0, int tile0, int tile1,
                                            int stride0, int pad_iv, int pad_amt) {
  uint64_t ga = (uint64_t)(uintptr_t)gaddr;
  v4u g0;
  g0[0] = 1u;                                     // count=1, user mode, no gather
  g0[1] = (unsigned)(uintptr_t)laddr;             // lds_addr (bytes)
  g0[2] = (unsigned)(ga & 0xffffffffu);           // global_addr[31:0]
  g0[3] = (unsigned)((ga >> 32) & 0x01ffffffu) | 0x80000000u;  // addr[56:32] | type=2
  const int tensor_d1 = 1 << 20;                  // rows: never clips (tiles always in-bounds)
  v8i g1;
  g1[0] = (1 << 16) | (1 << 20) | (pad_iv << 22) | (pad_amt << 25);  // data_size=2B, pad_enable
  g1[1] = (tensor_d0 & 0xffff) << 16;
  g1[2] = (int)(((unsigned)tensor_d0 >> 16) & 0xffffu) | ((tensor_d1 & 0xffff) << 16);
  g1[3] = (int)(((unsigned)tensor_d1 >> 16) & 0xffffu) | ((unsigned)tile0 << 16);
  g1[4] = tile1 & 0xffff;                         // tile_dim1 | tile_dim2=0
  g1[5] = stride0;                                // tensor_dim0_stride[31:0] (elements)
  g1[6] = 0;
  g1[7] = 0;
  v4i zz = (v4i){0, 0, 0, 0};
  v8i z8 = (v8i){0, 0, 0, 0, 0, 0, 0, 0};
  __builtin_amdgcn_tensor_load_to_lds((v4i)g0, g1, zz, zz, z8, 0);
}
__device__ __forceinline__ void wait_tensor0() {
#if __has_builtin(__builtin_amdgcn_s_wait_tensorcnt)
  __builtin_amdgcn_s_wait_tensorcnt(0);
#else
  asm volatile("s_wait_tensorcnt 0" ::: "memory");
#endif
}
#else
#define HAVE_TDM 0
#endif

// ---------------- fp32 -> f16 weight convert ----------------
__global__ void f32_to_f16_kernel(const float* __restrict__ in, h16* __restrict__ out, int n) {
  int i = (blockIdx.x * 256 + threadIdx.x) * 4;
  if (i + 3 < n) {
    float4 v = *(const float4*)(in + i);
    v4h o = { (h16)v.x, (h16)v.y, (h16)v.z, (h16)v.w };
    *(v4h*)(out + i) = o;
  }
}

// ---------------- embedding + sinusoidal PE ----------------
__global__ void embed_kernel(const int* __restrict__ tok, const float* __restrict__ emb,
                             float* __restrict__ xf, h16* __restrict__ xh) {
  int row = blockIdx.x;            // 0..M_-1
  int pos = row & (N_ - 1);        // position within sequence
  int t = threadIdx.x;
  int token = tok[row];
  const float scale = 22.62741699796952f;            // sqrt(512)
  const float nlog = -9.210340371976184f / 256.0f;   // -ln(10000)/depth
#pragma unroll
  for (int s = 0; s < 2; ++s) {
    int j = t + s * 256;
    float ang = (float)pos * expf((float)(j & 255) * nlog);
    float pe = (j < 256) ? sinf(ang) : cosf(ang);
    float v = emb[(size_t)token * D_ + j] * scale + pe;
    xf[(size_t)row * D_ + j] = v;
    xh[(size_t)row * D_ + j] = (h16)v;
  }
}

// ---------------- WMMA GEMM: C[M,Nn] = epilogue(A[M,K] @ W[K,Nn] + bias) ----------------
// MODE 0: store f16            (V projection)
// MODE 1: store f16 ELU        (Q/K projection, FFN1)
// MODE 2: store f32 + residual (O projection, FFN2)
#define BM 128
#define BN 128
#define BK 32
#define AS_STRIDE 40    // 32 + 8 pad (halves) = 16+4 DWORD rows  (TDM pad_iv=3, pad_amt=3)
#define BS_STRIDE 136   // 128 + 8 pad (halves) = 64+4 DWORD rows (TDM pad_iv=5, pad_amt=3)

template <int MODE>
__global__ __launch_bounds__(256, 2) void gemm16(
    const h16* __restrict__ A, const h16* __restrict__ W,
    const float* __restrict__ bias, const float* __restrict__ res,
    h16* __restrict__ outh, float* __restrict__ outf,
    int K, int Nn)
{
  __shared__ alignas(16) h16 As[2][BM * AS_STRIDE];
  __shared__ alignas(16) h16 Bs[2][BK * BS_STRIDE];

  const int tid  = threadIdx.x;
  const int lane = tid & 31;
  const int wave = tid >> 5;
  const int wm   = wave >> 1;        // 0..3 : 32-row slab
  const int wn   = wave & 1;         // 0..1 : 64-col slab
  const int m0   = blockIdx.y * BM;
  const int n0   = blockIdx.x * BN;

#if HAVE_TDM
  const h16* Ab = A + (size_t)m0 * K;   // advance by kt*BK elements along rows
  const h16* Bb = W + n0;               // advance by kt*BK*Nn elements down rows
  if (wave == 0) {  // one TDM DMA per tile: whole 2D tile incl. LDS padding
    tdm_load_2d(Ab,  &As[0][0], K,  BK, BM, K,  3, 3);
    tdm_load_2d(Bb,  &Bs[0][0], Nn, BN, BK, Nn, 5, 3);
    wait_tensor0();
  }
  __syncthreads();
#else
  // global->LDS staging assignment (16 halves / thread / tile)
  const int ar = tid >> 1;           // A row   0..127
  const int ac = (tid & 1) * 16;     // A col   0 / 16
  const int bk = tid >> 3;           // B row   0..31
  const int bc = (tid & 7) * 16;     // B col   0..112
  const h16* aptr = A + (size_t)(m0 + ar) * K + ac;
  const h16* bptr = W + (size_t)bk * Nn + n0 + bc;
#if HAVE_ASYNC_LDS
  async_cp16B(aptr,     &As[0][ar * AS_STRIDE + ac]);
  async_cp16B(aptr + 8, &As[0][ar * AS_STRIDE + ac + 8]);
  async_cp16B(bptr,     &Bs[0][bk * BS_STRIDE + bc]);
  async_cp16B(bptr + 8, &Bs[0][bk * BS_STRIDE + bc + 8]);
  wait_async0();
  __syncthreads();
#else
  uint4 arg0, arg1, brg0, brg1;
  arg0 = *(const uint4*)(aptr);
  arg1 = *(const uint4*)(aptr + 8);
  brg0 = *(const uint4*)(bptr);
  brg1 = *(const uint4*)(bptr + 8);
  *(uint4*)&As[0][ar * AS_STRIDE + ac]     = arg0;
  *(uint4*)&As[0][ar * AS_STRIDE + ac + 8] = arg1;
  *(uint4*)&Bs[0][bk * BS_STRIDE + bc]     = brg0;
  *(uint4*)&Bs[0][bk * BS_STRIDE + bc + 8] = brg1;
  __syncthreads();
#endif
#endif

  v8f acc[2][4];
#pragma unroll
  for (int mt = 0; mt < 2; ++mt)
#pragma unroll
    for (int nt = 0; nt < 4; ++nt) acc[mt][nt] = (v8f){};

  const int nk   = K >> 5;
  const int lrow = lane & 15;
  const int kof  = (lane >> 4) << 3;   // 0 or 8: A-operand K-half split per ISA layout

  for (int kt = 0; kt < nk; ++kt) {
    const int cur = kt & 1;
    // ---- issue next tile's staging (overlaps with WMMA work below) ----
#if HAVE_TDM
    if (kt + 1 < nk && wave == 0) {
      const int nxt = cur ^ 1;
      tdm_load_2d(Ab + (kt + 1) * BK,               &As[nxt][0], K,  BK, BM, K,  3, 3);
      tdm_load_2d(Bb + (size_t)(kt + 1) * BK * Nn,  &Bs[nxt][0], Nn, BN, BK, Nn, 5, 3);
    }
#elif HAVE_ASYNC_LDS
    if (kt + 1 < nk) {
      const int nxt = cur ^ 1;
      const h16* ap = aptr + (kt + 1) * BK;
      const h16* bp = bptr + (size_t)(kt + 1) * BK * Nn;
      async_cp16B(ap,     &As[nxt][ar * AS_STRIDE + ac]);
      async_cp16B(ap + 8, &As[nxt][ar * AS_STRIDE + ac + 8]);
      async_cp16B(bp,     &Bs[nxt][bk * BS_STRIDE + bc]);
      async_cp16B(bp + 8, &Bs[nxt][bk * BS_STRIDE + bc + 8]);
    }
#else
    if (kt + 1 < nk) {
      const h16* ap = aptr + (kt + 1) * BK;
      const h16* bp = bptr + (size_t)(kt + 1) * BK * Nn;
      arg0 = *(const uint4*)(ap);
      arg1 = *(const uint4*)(ap + 8);
      brg0 = *(const uint4*)(bp);
      brg1 = *(const uint4*)(bp + 8);
    }
#endif
    // ---- fragments + 8 WMMA ----
    v16h af[2];
#pragma unroll
    for (int mt = 0; mt < 2; ++mt) {
      int r = wm * 32 + mt * 16 + lrow;
      v8h lo = *(const v8h*)&As[cur][r * AS_STRIDE + kof];        // K 0..7 / 8..15
      v8h hi = *(const v8h*)&As[cur][r * AS_STRIDE + 16 + kof];   // K 16..23 / 24..31
      af[mt] = cat16(lo, hi);
    }
#pragma unroll
    for (int nt = 0; nt < 4; ++nt) {
      int c = wn * 64 + nt * 16;
      v16h bf;
#pragma unroll
      for (int s = 0; s < 16; ++s)
        bf[s] = Bs[cur][((s & 8) * 2 + kof + (s & 7)) * BS_STRIDE + c + lrow];
#pragma unroll
      for (int mt = 0; mt < 2; ++mt)
        acc[mt][nt] = wmma16(af[mt], bf, acc[mt][nt]);
    }
    // ---- commit next tile ----
    if (kt + 1 < nk) {
#if HAVE_TDM
      if (wave == 0) wait_tensor0();
#elif HAVE_ASYNC_LDS
      wait_async0();
#else
      const int nxt = cur ^ 1;
      *(uint4*)&As[nxt][ar * AS_STRIDE + ac]     = arg0;
      *(uint4*)&As[nxt][ar * AS_STRIDE + ac + 8] = arg1;
      *(uint4*)&Bs[nxt][bk * BS_STRIDE + bc]     = brg0;
      *(uint4*)&Bs[nxt][bk * BS_STRIDE + bc + 8] = brg1;
#endif
      __syncthreads();
    }
  }

  // epilogue (C layout: vgpr i -> row i (+8 for hi lanes), lane%16 -> col)
#pragma unroll
  for (int mt = 0; mt < 2; ++mt) {
    int rbase = m0 + wm * 32 + mt * 16 + ((lane >> 4) << 3);
#pragma unroll
    for (int nt = 0; nt < 4; ++nt) {
      int c = n0 + wn * 64 + nt * 16 + lrow;
      float bv = bias[c];
#pragma unroll
      for (int i = 0; i < 8; ++i) {
        int r = rbase + i;
        float v = acc[mt][nt][i] + bv;
        if constexpr (MODE == 1) v = elu1(v);
        if constexpr (MODE <= 1) outh[(size_t)r * Nn + c] = (h16)v;
        else                     outf[(size_t)r * Nn + c] = v + res[(size_t)r * Nn + c];
      }
    }
  }
}

// ---------------- ksum[b,h,d] = sum_n k[b,h,n,d] ----------------
__global__ void ksum_kernel(const h16* __restrict__ kh, float* __restrict__ ksum) {
  int bh = blockIdx.x; int b = bh >> 3, h = bh & 7;
  int t = threadIdx.x;
  int d = t & 63, chunk = t >> 6;     // 4 chunks of 1024 rows
  const h16* base = kh + (size_t)b * N_ * D_ + h * HD_ + d;
  float s = 0.0f;
  for (int n = chunk * 1024; n < chunk * 1024 + 1024; ++n)
    s += (float)base[(size_t)n * D_];
  __shared__ float red[256];
  red[t] = s;
  __syncthreads();
  if (chunk == 0)
    ksum[bh * HD_ + d] = red[d] + red[d + 64] + red[d + 128] + red[d + 192];
}

// ---------------- KV[b,h,d,e] = sum_n k[n,d] v[n,e]  (split-K over grid.y) ----------------
#define KVS 8
#define KT_STRIDE 40   // 32 n-cols + 8 pad (halves)
__global__ __launch_bounds__(256) void kv_kernel(const h16* __restrict__ kh,
                                                 const h16* __restrict__ vh,
                                                 float* __restrict__ kvpart) {
  int bh = blockIdx.x; int b = bh >> 3, h = bh & 7;
  int split = blockIdx.y;
  int tid = threadIdx.x, lane = tid & 31, wave = tid >> 5;
  int mt = wave >> 1;            // d tile 0..3
  int np = (wave & 1) * 2;       // e tile pair base 0 / 2
  __shared__ alignas(16) h16 Kt[64][KT_STRIDE];   // k chunk stored TRANSPOSED: Kt[d][n]
  __shared__ alignas(16) h16 Vs[32][72];
  v8f acc0 = (v8f){}, acc1 = (v8f){};
  int lrow = lane & 15, kof = (lane >> 4) << 3;
  const int lr = tid >> 3;           // staging row (n) 0..31
  const int lc = (tid & 7) * 8;      // staging col (d) 0..56
  int n_begin = split * (N_ / KVS);
  for (int nc = n_begin; nc < n_begin + N_ / KVS; nc += 32) {
    size_t g = (size_t)(b * N_ + nc + lr) * D_ + h * HD_ + lc;
#if HAVE_ASYNC_LDS
    async_cp16B(vh + g, &Vs[lr][lc]);
#else
    *(uint4*)&Vs[lr][lc] = *(const uint4*)(vh + g);
#endif
    union { uint4 u; h16 e[8]; } kk;
    kk.u = *(const uint4*)(kh + g);
#pragma unroll
    for (int j = 0; j < 8; ++j) Kt[lc + j][lr] = kk.e[j];   // transpose on store
#if HAVE_ASYNC_LDS
    wait_async0();
#endif
    __syncthreads();
    // A = k^T : clean pair of b128 LDS loads (same shape as main GEMM)
    int d = mt * 16 + lrow;
    v8h alo = *(const v8h*)&Kt[d][kof];
    v8h ahi = *(const v8h*)&Kt[d][16 + kof];
    v16h af = cat16(alo, ahi);
    v16h bf0, bf1;
#pragma unroll
    for (int s = 0; s < 16; ++s) {
      int kk2 = (s & 8) * 2 + kof + (s & 7);
      bf0[s] = Vs[kk2][np * 16 + lrow];
      bf1[s] = Vs[kk2][np * 16 + 16 + lrow];
    }
    acc0 = wmma16(af, bf0, acc0);
    acc1 = wmma16(af, bf1, acc1);
    __syncthreads();
  }
  int rbase = mt * 16 + ((lane >> 4) << 3);
  size_t pb = ((size_t)bh * KVS + split) * 4096;
#pragma unroll
  for (int i = 0; i < 8; ++i) {
    int r = rbase + i;
    kvpart[pb + (size_t)r * 64 + np * 16 + lrow]        = acc0[i];
    kvpart[pb + (size_t)r * 64 + (np + 1) * 16 + lrow]  = acc1[i];
  }
}

__global__ void kv_reduce_kernel(const float* __restrict__ kvpart, h16* __restrict__ kvh) {
  int i = blockIdx.x * 256 + threadIdx.x;  // over B_*H_*64*64 = 131072
  int bh = i >> 12, off = i & 4095;
  float s = 0.0f;
#pragma unroll
  for (int sp = 0; sp < KVS; ++sp) s += kvpart[((size_t)bh * KVS + sp) * 4096 + off];
  kvh[i] = (h16)s;
}

// ---------------- o[n,e] = (q[n,:] @ KV) / (q[n,:] . ksum) ----------------
__global__ __launch_bounds__(256) void attn_apply_kernel(const h16* __restrict__ qh,
                                                         const h16* __restrict__ kvh,
                                                         const float* __restrict__ ksum,
                                                         h16* __restrict__ oh) {
  int bh = blockIdx.x; int b = bh >> 3, h = bh & 7;
  int n0 = blockIdx.y * 128;
  int tid = threadIdx.x, lane = tid & 31, wave = tid >> 5;
  __shared__ float denom[128];
  __shared__ float dpart[256];
  __shared__ alignas(16) h16 KVs[64][72];
  {  // stage KV (64x64 f16) once per block
    int r = tid >> 2;
    int c = (tid & 3) * 16;
    const h16* kp = kvh + ((size_t)bh * 64 + r) * 64 + c;
    *(uint4*)&KVs[r][c]     = *(const uint4*)(kp);
    *(uint4*)&KVs[r][c + 8] = *(const uint4*)(kp + 8);
  }
  {  // per-row denominator q . ksum
    int r = tid >> 1, half = tid & 1;
    const h16* qp = qh + (size_t)(b * N_ + n0 + r) * D_ + h * HD_ + half * 32;
    const float* ks = ksum + bh * HD_ + half * 32;
    float s = 0.0f;
#pragma unroll
    for (int d2 = 0; d2 < 32; ++d2) s += (float)qp[d2] * ks[d2];
    dpart[tid] = s;
  }
  __syncthreads();
  if ((tid & 1) == 0) denom[tid >> 1] = dpart[tid] + dpart[tid + 1];
  __syncthreads();

  int lrow = lane & 15, kof = (lane >> 4) << 3;
  int rw = wave * 16;
  v8f acc[4];
#pragma unroll
  for (int nt = 0; nt < 4; ++nt) acc[nt] = (v8f){};
#pragma unroll
  for (int kc = 0; kc < 64; kc += 32) {
    const h16* qp = qh + (size_t)(b * N_ + n0 + rw + lrow) * D_ + h * HD_ + kc + kof;
    v8h lo = *(const v8h*)(qp);
    v8h hi = *(const v8h*)(qp + 16);
    v16h af = cat16(lo, hi);
#pragma unroll
    for (int nt = 0; nt < 4; ++nt) {
      v16h bf;
#pragma unroll
      for (int s = 0; s < 16; ++s)
        bf[s] = KVs[kc + (s & 8) * 2 + kof + (s & 7)][nt * 16 + lrow];
      acc[nt] = wmma16(af, bf, acc[nt]);
    }
  }
#pragma unroll
  for (int nt = 0; nt < 4; ++nt) {
    int c = nt * 16 + lrow;
#pragma unroll
    for (int i = 0; i < 8; ++i) {
      int r = rw + i + ((lane >> 4) << 3);
      float v = acc[nt][i] / denom[r];
      oh[(size_t)(b * N_ + n0 + r) * D_ + h * HD_ + c] = (h16)v;
    }
  }
}

// ---------------- LayerNorm over D=512; writes fp32 master + f16 shadow ----------------
__global__ void ln_kernel(const float* __restrict__ y, const float* __restrict__ g,
                          const float* __restrict__ bta, float* __restrict__ xf,
                          h16* __restrict__ xh) {
  int row = blockIdx.x, t = threadIdx.x;
  const float* yr = y + (size_t)row * D_;
  float v0 = yr[t], v1 = yr[t + 256];
  __shared__ float red[256];
  red[t] = v0 + v1;
  __syncthreads();
#pragma unroll
  for (int s = 128; s > 0; s >>= 1) { if (t < s) red[t] += red[t + s]; __syncthreads(); }
  float mean = red[0] * (1.0f / D_);
  __syncthreads();
  float d0 = v0 - mean, d1 = v1 - mean;
  red[t] = d0 * d0 + d1 * d1;
  __syncthreads();
#pragma unroll
  for (int s = 128; s > 0; s >>= 1) { if (t < s) red[t] += red[t + s]; __syncthreads(); }
  float rstd = rsqrtf(red[0] * (1.0f / D_) + 1e-3f);
  float o0 = d0 * rstd * g[t] + bta[t];
  float o1 = d1 * rstd * g[t + 256] + bta[t + 256];
  xf[(size_t)row * D_ + t] = o0;         xh[(size_t)row * D_ + t] = (h16)o0;
  xf[(size_t)row * D_ + t + 256] = o1;   xh[(size_t)row * D_ + t + 256] = (h16)o1;
}

// ---------------- host launcher ----------------
extern "C" void kernel_launch(void* const* d_in, const int* in_sizes, int n_in,
                              void* d_out, int out_size, void* d_ws, size_t ws_size,
                              hipStream_t stream) {
  (void)in_sizes; (void)n_in; (void)out_size; (void)ws_size;
  const int*   tokens = (const int*)  d_in[0];
  const float* emb    = (const float*)d_in[1];
  const float* Wq     = (const float*)d_in[2];
  const float* bq     = (const float*)d_in[3];
  const float* Wk     = (const float*)d_in[4];
  const float* bk     = (const float*)d_in[5];
  const float* Wv     = (const float*)d_in[6];
  const float* bv     = (const float*)d_in[7];
  const float* Wo     = (const float*)d_in[8];
  const float* bo     = (const float*)d_in[9];
  const float* ln1g   = (const float*)d_in[10];
  const float* ln1b   = (const float*)d_in[11];
  const float* W1     = (const float*)d_in[12];
  const float* b1     = (const float*)d_in[13];
  const float* W2     = (const float*)d_in[14];
  const float* b2     = (const float*)d_in[15];
  const float* ln2g   = (const float*)d_in[16];
  const float* ln2b   = (const float*)d_in[17];

  char* ws = (char*)d_ws;
  size_t off = 0;
  auto alloc = [&](size_t bytes) -> void* {
    void* p = ws + off;
    off += (bytes + 255) & ~(size_t)255;
    return p;
  };
  float* xf     = (float*)alloc((size_t)M_ * D_ * 4);
  float* yf     = (float*)alloc((size_t)M_ * D_ * 4);
  h16*   xh     = (h16*)  alloc((size_t)M_ * D_ * 2);
  h16*   qh     = (h16*)  alloc((size_t)M_ * D_ * 2);
  h16*   kh     = (h16*)  alloc((size_t)M_ * D_ * 2);
  h16*   vh     = (h16*)  alloc((size_t)M_ * D_ * 2);
  h16*   ohb    = (h16*)  alloc((size_t)M_ * D_ * 2);
  h16*   h1h    = (h16*)  alloc((size_t)M_ * DFF_ * 2);
  h16*   kvh    = (h16*)  alloc((size_t)B_ * H_ * 64 * 64 * 2);
  float* kvpart = (float*)alloc((size_t)B_ * H_ * KVS * 64 * 64 * 4);
  float* ksumb  = (float*)alloc((size_t)B_ * H_ * HD_ * 4);
  h16*   Wqh    = (h16*)  alloc((size_t)L_ * D_ * D_ * 2);
  h16*   Wkh    = (h16*)  alloc((size_t)L_ * D_ * D_ * 2);
  h16*   Wvh    = (h16*)  alloc((size_t)L_ * D_ * D_ * 2);
  h16*   Woh    = (h16*)  alloc((size_t)L_ * D_ * D_ * 2);
  h16*   W1h    = (h16*)  alloc((size_t)L_ * D_ * DFF_ * 2);
  h16*   W2h    = (h16*)  alloc((size_t)L_ * DFF_ * D_ * 2);

  // weight conversion (fp32 -> f16), done every call (deterministic)
  int nDD = L_ * D_ * D_, nDF = L_ * D_ * DFF_;
  f32_to_f16_kernel<<<nDD / 1024, 256, 0, stream>>>(Wq, Wqh, nDD);
  f32_to_f16_kernel<<<nDD / 1024, 256, 0, stream>>>(Wk, Wkh, nDD);
  f32_to_f16_kernel<<<nDD / 1024, 256, 0, stream>>>(Wv, Wvh, nDD);
  f32_to_f16_kernel<<<nDD / 1024, 256, 0, stream>>>(Wo, Woh, nDD);
  f32_to_f16_kernel<<<nDF / 1024, 256, 0, stream>>>(W1, W1h, nDF);
  f32_to_f16_kernel<<<nDF / 1024, 256, 0, stream>>>(W2, W2h, nDF);

  embed_kernel<<<M_, 256, 0, stream>>>(tokens, emb, xf, xh);

  dim3 g512(D_ / BN, M_ / BM);     // (4, 128)
  dim3 g2048(DFF_ / BN, M_ / BM);  // (16, 128)

  for (int l = 0; l < L_; ++l) {
    const size_t wDD = (size_t)l * D_ * D_;
    gemm16<1><<<g512, 256, 0, stream>>>(xh, Wqh + wDD, bq + l * D_, nullptr, qh, nullptr, D_, D_);
    gemm16<1><<<g512, 256, 0, stream>>>(xh, Wkh + wDD, bk + l * D_, nullptr, kh, nullptr, D_, D_);
    gemm16<0><<<g512, 256, 0, stream>>>(xh, Wvh + wDD, bv + l * D_, nullptr, vh, nullptr, D_, D_);

    ksum_kernel<<<B_ * H_, 256, 0, stream>>>(kh, ksumb);
    kv_kernel<<<dim3(B_ * H_, KVS), 256, 0, stream>>>(kh, vh, kvpart);
    kv_reduce_kernel<<<(B_ * H_ * 4096) / 256, 256, 0, stream>>>(kvpart, kvh);
    attn_apply_kernel<<<dim3(B_ * H_, N_ / 128), 256, 0, stream>>>(qh, kvh, ksumb, ohb);

    gemm16<2><<<g512, 256, 0, stream>>>(ohb, Woh + wDD, bo + l * D_, xf, nullptr, yf, D_, D_);
    ln_kernel<<<M_, 256, 0, stream>>>(yf, ln1g + l * D_, ln1b + l * D_, xf, xh);

    gemm16<1><<<g2048, 256, 0, stream>>>(xh, W1h + (size_t)l * D_ * DFF_, b1 + l * DFF_,
                                         nullptr, h1h, nullptr, D_, DFF_);
    gemm16<2><<<g512, 256, 0, stream>>>(h1h, W2h + (size_t)l * DFF_ * D_, b2 + l * D_,
                                        xf, nullptr, yf, DFF_, D_);
    ln_kernel<<<M_, 256, 0, stream>>>(yf, ln2g + l * D_, ln2b + l * D_, xf, xh);
  }

  (void)hipMemcpyAsync(d_out, xf, (size_t)M_ * D_ * 4, hipMemcpyDeviceToDevice, stream);
}
